// Optimize_geodesicShooting_31207232372844
// MI455X (gfx1250) — matrix operand order repaired and checked by reading, not compiled, not run on hardware
//
#include <hip/hip_runtime.h>

// ---------------------------------------------------------------------------
// Metamorphosis geodesic shooting, 20 steps, 2048x2048 fp32 img/z.
// Separable 25-tap Gaussian blur as banded matmul on the CDNA5 WMMA pipe
// (v_wmma_f32_16x16x32_f16, 2 ops per 16x16 tile per pass), with branch-free
// fragment packing from packed-f16 LDS tiles and a zero-padded Toeplitz table.
// Intermediate planes (mom/tmp/v) are f16 in HBM to halve blur traffic.
// ---------------------------------------------------------------------------

#define HH 2048
#define WW 2048
#define NSTEP 20
#define DT (1.0f / 20.0f)
#define MUC 1.0f
#define RHOC 1.0f
#define SIG 4.0f
#define RAD 12      // ceil(3*sigma)
#define KLEN 25

typedef __attribute__((ext_vector_type(16))) _Float16     v16h;
typedef __attribute__((ext_vector_type(8)))  float        v8f;
typedef __attribute__((ext_vector_type(8)))  unsigned int v8u;

union hpack { _Float16 h[2]; unsigned int u; };
__device__ __forceinline__ unsigned int pack2(float a, float b) {
  hpack p; p.h[0] = (_Float16)a; p.h[1] = (_Float16)b; return p.u;
}

// ---- central differences, one-sided at borders (matches reference cdiff) ----
__device__ __forceinline__ float cdH(const float* __restrict__ x, int r, int c) {
  if (r == 0)      return x[WW + c] - x[c];
  if (r == HH - 1) return x[(HH-1)*WW + c] - x[(HH-2)*WW + c];
  return 0.5f * (x[(r+1)*WW + c] - x[(r-1)*WW + c]);
}
__device__ __forceinline__ float cdW(const float* __restrict__ x, int r, int c) {
  if (c == 0)      return x[r*WW + 1] - x[r*WW];
  if (c == WW - 1) return x[r*WW + WW-1] - x[r*WW + WW-2];
  return 0.5f * (x[r*WW + c+1] - x[r*WW + c-1]);
}
__device__ __forceinline__ float pq(const float* __restrict__ a,
                                    const _Float16* __restrict__ b, int r, int c) {
  int i = r*WW + c; return a[i] * (float)b[i];
}
__device__ __forceinline__ float cdHp(const float* __restrict__ a,
                                      const _Float16* __restrict__ b, int r, int c) {
  if (r == 0)      return pq(a,b,1,c)    - pq(a,b,0,c);
  if (r == HH - 1) return pq(a,b,HH-1,c) - pq(a,b,HH-2,c);
  return 0.5f * (pq(a,b,r+1,c) - pq(a,b,r-1,c));
}
__device__ __forceinline__ float cdWp(const float* __restrict__ a,
                                      const _Float16* __restrict__ b, int r, int c) {
  if (c == 0)      return pq(a,b,r,1)    - pq(a,b,r,0);
  if (c == WW - 1) return pq(a,b,r,WW-1) - pq(a,b,r,WW-2);
  return 0.5f * (pq(a,b,r,c+1) - pq(a,b,r,c-1));
}

// ---- zero-padded, scaled Gaussian table in LDS: kxh[t+16] = scale*k[t] ----
// Index range used by packing: (p - m) + 16 in [1, 79]  -> size 80, no guards.
#define KPAD 16
#define KXN  80
__device__ __forceinline__ void fill_kxh(_Float16* kxh, float* inv, int tid, float scale) {
  if (tid == 0) {
    float s = 0.f;
    #pragma unroll
    for (int i = 0; i < KLEN; ++i) {
      float d = (float)(i - RAD);
      s += __expf(-d*d / (2.f * SIG * SIG));
    }
    *inv = scale / s;
  }
  __syncthreads();
  if (tid < KXN) {
    int t = tid - KPAD;
    float v = 0.f;
    if (t >= 0 && t < KLEN) {
      float d = (float)(t - RAD);
      v = __expf(-d*d / (2.f * SIG * SIG)) * (*inv);
    }
    kxh[tid] = (_Float16)v;
  }
}

// ---------------------------------------------------------------------------
// K1: mom = -z * grad(img)   -> two f16 planes (d/dH, d/dW)
// ---------------------------------------------------------------------------
__global__ void __launch_bounds__(256)
k_mom(const float* __restrict__ img, const float* __restrict__ z,
      _Float16* __restrict__ mom0, _Float16* __restrict__ mom1) {
  int c = blockIdx.x * blockDim.x + threadIdx.x;
  int r = blockIdx.y * blockDim.y + threadIdx.y;
  int i = r * WW + c;
  float zz = z[i];
  mom0[i] = (_Float16)(-zz * cdH(img, r, c));
  mom1[i] = (_Float16)(-zz * cdW(img, r, c));
}

// ---------------------------------------------------------------------------
// K2: vertical 25-tap blur via WMMA. Block = 8 waves, 128 rows x 16 cols.
// out(16x16) = Toeplitz(16x64) x data(64x16).
// LDS tile stored transposed as packed row-pairs: tile32[col][rowpair].
// ---------------------------------------------------------------------------
#define NPAIR 88          // ceil(176/2): full 64-window span for wave 7
__global__ void __launch_bounds__(256)
k_blur_v(const _Float16* __restrict__ src, _Float16* __restrict__ dst, float scale) {
  __shared__ unsigned int tile32[16][NPAIR + 1];
  __shared__ _Float16 kxh[KXN];
  __shared__ float inv;
  const int tid = threadIdx.x;
  const int plane = blockIdx.z * (HH * WW);
  src += plane; dst += plane;
  const int c0 = blockIdx.x * 16;
  const int r0 = blockIdx.y * 128;

  fill_kxh(kxh, &inv, tid, scale);

  // stage (rows r0-12 .. r0+163, 16 cols) as packed row-pairs, zero border
  for (int i = tid; i < 16 * NPAIR; i += 256) {
    int pi = i >> 4, n = i & 15;
    int gr0 = r0 - RAD + 2*pi;
    float lo = (gr0 >= 0     && gr0 < HH)     ? (float)src[gr0*WW + c0 + n]     : 0.f;
    float hi = (gr0+1 >= 0   && gr0+1 < HH)   ? (float)src[(gr0+1)*WW + c0 + n] : 0.f;
    tile32[n][pi] = pack2(lo, hi);
  }
  __syncthreads();

  const int lane = tid & 31, w = tid >> 5;
  const int m = lane & 15, half = lane >> 4;
  const int n = m;

  v8f acc = {};
  #pragma unroll
  for (int f = 0; f < 2; ++f) {
    v16h a; v8u bu;
    #pragma unroll
    for (int v = 0; v < 8; ++v) {
      // A (Toeplitz, 16-bit 16x32 layout): lane row m, K-pairs per VGPR
      int k  = 16*half + 2*v;
      int p0 = 32*f + k;
      a[2*v]   = kxh[p0     - m + KPAD];
      a[2*v+1] = kxh[p0 + 1 - m + KPAD];
      // B (data, 16-bit 32x16 layout): lane col n, row-pair per VGPR
      bu[v] = tile32[n][16*f + 8*half + v + 8*w];
    }
    v16h b = __builtin_bit_cast(v16h, bu);
    acc = __builtin_amdgcn_wmma_f32_16x16x32_f16(false, a, false, b,
                                                 (short)0, acc, false, false);
  }

  const int rbase = r0 + 16*w;
  #pragma unroll
  for (int j = 0; j < 8; ++j)
    dst[(rbase + j + 8*half) * WW + c0 + n] = (_Float16)acc[j];
}

// ---------------------------------------------------------------------------
// K3: horizontal 25-tap blur via WMMA. Block = 8 waves, 16 rows x 128 cols.
// out(16x16) = data(16x64) x Toeplitz(64x16).
// LDS tile stored as packed col-pairs: tile32[row][colpair].
// ---------------------------------------------------------------------------
__global__ void __launch_bounds__(256)
k_blur_h(const _Float16* __restrict__ src, _Float16* __restrict__ dst, float scale) {
  __shared__ unsigned int tile32[16][NPAIR + 1];
  __shared__ _Float16 kxh[KXN];
  __shared__ float inv;
  const int tid = threadIdx.x;
  const int plane = blockIdx.z * (HH * WW);
  src += plane; dst += plane;
  const int c0 = blockIdx.x * 128;
  const int r0 = blockIdx.y * 16;

  fill_kxh(kxh, &inv, tid, scale);

  // stage (16 rows, cols c0-12 .. c0+163) as packed col-pairs, zero border.
  // pairs are 4B-aligned in global (c0-12 even), so interior is one b32 load.
  for (int i = tid; i < 16 * NPAIR; i += 256) {
    int rr = i / NPAIR, ci = i % NPAIR;
    int gc0 = c0 - RAD + 2*ci;
    unsigned int u;
    if (gc0 >= 0 && gc0 + 1 < WW) {
      u = *(const unsigned int*)(src + (size_t)rr*WW + (size_t)r0*WW + gc0);
    } else {
      float lo = (gc0 >= 0   && gc0 < WW)   ? (float)src[(r0+rr)*WW + gc0]   : 0.f;
      float hi = (gc0+1 >= 0 && gc0+1 < WW) ? (float)src[(r0+rr)*WW + gc0+1] : 0.f;
      u = pack2(lo, hi);
    }
    tile32[rr][ci] = u;
  }
  __syncthreads();

  const int lane = tid & 31, w = tid >> 5;
  const int m = lane & 15, half = lane >> 4;
  const int n = m;

  v8f acc = {};
  #pragma unroll
  for (int f = 0; f < 2; ++f) {
    v8u au; v16h b;
    #pragma unroll
    for (int v = 0; v < 8; ++v) {
      // A (data, 16-bit 16x32 layout): lane row m, K-pairs per VGPR
      int k  = 16*half + 2*v;
      au[v] = tile32[m][(32*f + k) / 2 + 8*w];
      // B (Toeplitz, 16-bit 32x16 layout): lane col n, row-pair per VGPR
      int q0 = 32*f + 16*half + 2*v;
      b[2*v]   = kxh[q0     - n + KPAD];
      b[2*v+1] = kxh[q0 + 1 - n + KPAD];
    }
    v16h a = __builtin_bit_cast(v16h, au);
    acc = __builtin_amdgcn_wmma_f32_16x16x32_f16(false, a, false, b,
                                                 (short)0, acc, false, false);
  }

  #pragma unroll
  for (int j = 0; j < 8; ++j)
    dst[(r0 + j + 8*half) * WW + c0 + 16*w + n] = (_Float16)acc[j];
}

// ---------------------------------------------------------------------------
// K4: img' = img - (gradI . v) dt + mu z dt ;  z' = z - div(z v) dt
// ---------------------------------------------------------------------------
__global__ void __launch_bounds__(256)
k_update(const float* __restrict__ img, const float* __restrict__ z,
         const _Float16* __restrict__ v0, const _Float16* __restrict__ v1,
         float* __restrict__ img_out,     float* __restrict__ z_out) {
  int c = blockIdx.x * blockDim.x + threadIdx.x;
  int r = blockIdx.y * blockDim.y + threadIdx.y;
  int i = r * WW + c;
  float zz = z[i];
  float adv = cdH(img, r, c) * (float)v0[i] + cdW(img, r, c) * (float)v1[i];
  img_out[i] = img[i] - adv * DT + zz * (MUC * DT);
  float div = cdHp(z, v0, r, c) + cdWp(z, v1, r, c);
  z_out[i] = zz - div * DT;
}

// ---------------------------------------------------------------------------
extern "C" void kernel_launch(void* const* d_in, const int* in_sizes, int n_in,
                              void* d_out, int out_size, void* d_ws, size_t ws_size,
                              hipStream_t stream) {
  const float* img_in = (const float*)d_in[0];
  const float* z_in   = (const float*)d_in[1];
  float* out = (float*)d_out;
  float* ws  = (float*)d_ws;
  const size_t N = (size_t)HH * WW;

  // fp32 planes: img ping/pong, z ping/pong.  f16 planes: mom[2] (also v), tmp[2].
  float* imgb[2] = { ws,         ws + N };
  float* zb[2]   = { ws + 2*N,   ws + 3*N };
  _Float16* h16  = (_Float16*)(ws + 4*N);
  _Float16* mom  = h16;          // 2 contiguous planes; holds v after K3
  _Float16* tmp  = h16 + 2*N;    // 2 contiguous planes

  hipMemcpyAsync(imgb[0], img_in, N * sizeof(float), hipMemcpyDeviceToDevice, stream);
  hipMemcpyAsync(zb[0],   z_in,   N * sizeof(float), hipMemcpyDeviceToDevice, stream);

  dim3 bp(64, 4), gp(WW / 64, HH / 4);
  dim3 gbv(WW / 16, HH / 128, 2);   // vertical blur tiles
  dim3 gbh(WW / 128, HH / 16, 2);   // horizontal blur tiles

  int cur = 0;
  for (int s = 0; s < NSTEP; ++s) {
    k_mom   <<<gp,  bp,  0, stream>>>(imgb[cur], zb[cur], mom, mom + N);
    k_blur_v<<<gbv, 256, 0, stream>>>(mom, tmp, 1.0f);           // kh pass
    k_blur_h<<<gbh, 256, 0, stream>>>(tmp, mom, RHOC / MUC);     // kw pass -> v
    k_update<<<gp,  bp,  0, stream>>>(imgb[cur], zb[cur], mom, mom + N,
                                      imgb[1 - cur], zb[1 - cur]);
    cur ^= 1;
  }
  hipMemcpyAsync(out, imgb[cur], N * sizeof(float), hipMemcpyDeviceToDevice, stream);
}
